// SALoss_38989713113324
// MI455X (gfx1250) — matrix operand order repaired and checked by reading, not compiled, not run on hardware
//
#include <hip/hip_runtime.h>
#include <math.h>

// ---------------- problem constants (fixed by setup_inputs) ----------------
#define MC 16          // num classes
#define KD 32          // embedding dim
#define EPS 1e-8f

// ---------------- workspace layout (floats) ----------------
#define OFF_CNT    0      // 16  counts (accumulated as float)
#define OFF_SPTS   16     // 48  sum of points per class
#define OFF_SEMB   64     // 512 sum of embeddings per class
#define OFF_INTRA  576    // 16  per-class intra sums
#define OFF_MEN    592    // 16  ||mean_emb|| (clamped to EPS)
#define OFF_MEMB   608    // 512 mean_emb
#define OFF_MN     1120   // 512 normalized mean_emb, class0 zeroed
#define OFF_MPTS   1632   // 48  mean points
#define WS_ZERO    592    // accumulators that must start at zero

typedef __attribute__((ext_vector_type(16))) _Float16 v16h;
typedef __attribute__((ext_vector_type(8)))  float    v8f;

struct ll2 { long long x, y; };   // 16B label pair -> global_load_b128 friendly

// ---------------- kernel 0: zero accumulators ----------------
__global__ void k_zero(float* ws) {
    for (int i = threadIdx.x; i < WS_ZERO; i += blockDim.x) ws[i] = 0.0f;
}

// ---------------- kernel 1: segment sums (counts, pts, emb) ----------------
// 4 points per thread, all global traffic via b128 loads.
__global__ void k_pass1(const float* __restrict__ pts,
                        const float* __restrict__ emb,
                        const long long* __restrict__ seg,
                        float* __restrict__ ws, int N) {
    __shared__ float sbuf[16 + 48 + 512];   // cnt | spts | semb, same layout as ws base
    const int t = threadIdx.x;
    for (int i = t; i < 576; i += blockDim.x) sbuf[i] = 0.0f;
    __syncthreads();

    float* s_cnt = sbuf;
    float* s_pts = sbuf + 16;
    float* s_emb = sbuf + 64;

    const int gid     = blockIdx.x * blockDim.x + t;
    const int gstride = gridDim.x * blockDim.x;
    const int NV      = N >> 2;                     // groups of 4 points

    const float4* p4 = (const float4*)pts;          // 3 x float4 = 4 points
    const ll2*    s2 = (const ll2*)seg;             // 2 x ll2    = 4 labels

    for (int gi = gid; gi < NV; gi += gstride) {
        const ll2 sa = s2[(size_t)gi * 2];
        const ll2 sb = s2[(size_t)gi * 2 + 1];
        int c[4] = { (int)sa.x, (int)sa.y, (int)sb.x, (int)sb.y };

        const float4 q0 = p4[(size_t)gi * 3];       // x0 y0 z0 x1
        const float4 q1 = p4[(size_t)gi * 3 + 1];   // y1 z1 x2 y2
        const float4 q2 = p4[(size_t)gi * 3 + 2];   // z2 x3 y3 z3
        const float px[4] = { q0.x, q0.w, q1.z, q2.y };
        const float py[4] = { q0.y, q1.x, q1.w, q2.z };
        const float pz[4] = { q0.z, q1.y, q2.x, q2.w };

#pragma unroll
        for (int j = 0; j < 4; ++j) {
            atomicAdd(&s_cnt[c[j]], 1.0f);
            atomicAdd(&s_pts[c[j] * 3 + 0], px[j]);
            atomicAdd(&s_pts[c[j] * 3 + 1], py[j]);
            atomicAdd(&s_pts[c[j] * 3 + 2], pz[j]);
        }
#pragma unroll
        for (int k = 0; k < KD; ++k) {
            // embedding is (K, N): fixed k, 4 consecutive n -> one b128, coalesced
            const float4 e = ((const float4*)(emb + (size_t)k * N))[gi];
            atomicAdd(&s_emb[c[0] * KD + k], e.x);
            atomicAdd(&s_emb[c[1] * KD + k], e.y);
            atomicAdd(&s_emb[c[2] * KD + k], e.z);
            atomicAdd(&s_emb[c[3] * KD + k], e.w);
        }
    }
    // scalar tail (N not divisible by 4; no-op for N = 1M)
    for (int n = NV * 4 + gid; n < N; n += gstride) {
        const int c = (int)seg[n];
        atomicAdd(&s_cnt[c], 1.0f);
        atomicAdd(&s_pts[c * 3 + 0], pts[(size_t)n * 3 + 0]);
        atomicAdd(&s_pts[c * 3 + 1], pts[(size_t)n * 3 + 1]);
        atomicAdd(&s_pts[c * 3 + 2], pts[(size_t)n * 3 + 2]);
        for (int k = 0; k < KD; ++k)
            atomicAdd(&s_emb[c * KD + k], emb[(size_t)k * N + n]);
    }
    __syncthreads();
    for (int i = t; i < 576; i += blockDim.x) {
        float v = sbuf[i];
        if (v != 0.0f) atomicAdd(&ws[i], v);
    }
}

// ---------------- kernel 2: means, norms, normalized rows ----------------
__global__ void k_means(float* ws) {
    const int t = threadIdx.x;            // launched with 512 threads
    const float* cnt = ws + OFF_CNT;
    if (t < 512) {
        const int c = t / KD;
        ws[OFF_MEMB + t] = ws[OFF_SEMB + t] / fmaxf(cnt[c], 1.0f);
    }
    if (t < 48) {
        const int c = t / 3;
        ws[OFF_MPTS + t] = ws[OFF_SPTS + t] / fmaxf(cnt[c], 1.0f);
    }
    __syncthreads();
    if (t < MC) {
        float s = 0.0f;
#pragma unroll
        for (int k = 0; k < KD; ++k) {
            float v = ws[OFF_MEMB + t * KD + k];
            s += v * v;
        }
        ws[OFF_MEN + t] = fmaxf(sqrtf(s), EPS);
    }
    __syncthreads();
    if (t < 512) {
        const int c = t / KD;
        ws[OFF_MN + t] = (c == 0) ? 0.0f : ws[OFF_MEMB + t] / ws[OFF_MEN + c];
    }
}

// ---------------- kernel 3: per-point intra terms ----------------
// 4 points per thread, b128 loads; means/norms staged in LDS.
__global__ void k_pass2(const float* __restrict__ pts,
                        const float* __restrict__ emb,
                        const long long* __restrict__ seg,
                        float* __restrict__ ws, int N) {
    __shared__ float s_mpts[48];
    __shared__ float s_memb[512];
    __shared__ float s_men[16];
    __shared__ float s_intra[16];
    const int t = threadIdx.x;
    for (int i = t; i < 48;  i += blockDim.x) s_mpts[i] = ws[OFF_MPTS + i];
    for (int i = t; i < 512; i += blockDim.x) s_memb[i] = ws[OFF_MEMB + i];
    if (t < 16) { s_men[t] = ws[OFF_MEN + t]; s_intra[t] = 0.0f; }
    __syncthreads();

    const int gid     = blockIdx.x * blockDim.x + t;
    const int gstride = gridDim.x * blockDim.x;
    const int NV      = N >> 2;

    const float4* p4 = (const float4*)pts;
    const ll2*    s2 = (const ll2*)seg;

    for (int gi = gid; gi < NV; gi += gstride) {
        const ll2 sa = s2[(size_t)gi * 2];
        const ll2 sb = s2[(size_t)gi * 2 + 1];
        const int c[4] = { (int)sa.x, (int)sa.y, (int)sb.x, (int)sb.y };

        const float4 q0 = p4[(size_t)gi * 3];
        const float4 q1 = p4[(size_t)gi * 3 + 1];
        const float4 q2 = p4[(size_t)gi * 3 + 2];
        const float px[4] = { q0.x, q0.w, q1.z, q2.y };
        const float py[4] = { q0.y, q1.x, q1.w, q2.z };
        const float pz[4] = { q0.z, q1.y, q2.x, q2.w };

        float dot[4] = {0.f, 0.f, 0.f, 0.f};
        float nrm[4] = {0.f, 0.f, 0.f, 0.f};
#pragma unroll
        for (int k = 0; k < KD; ++k) {
            const float4 e = ((const float4*)(emb + (size_t)k * N))[gi];
            dot[0] += e.x * s_memb[c[0] * KD + k];  nrm[0] += e.x * e.x;
            dot[1] += e.y * s_memb[c[1] * KD + k];  nrm[1] += e.y * e.y;
            dot[2] += e.z * s_memb[c[2] * KD + k];  nrm[2] += e.z * e.z;
            dot[3] += e.w * s_memb[c[3] * KD + k];  nrm[3] += e.w * e.w;
        }
#pragma unroll
        for (int j = 0; j < 4; ++j) {
            const float dx = px[j] - s_mpts[c[j] * 3 + 0];
            const float dy = py[j] - s_mpts[c[j] * 3 + 1];
            const float dz = pz[j] - s_mpts[c[j] * 3 + 2];
            const float d  = sqrtf(dx * dx + dy * dy + dz * dz);
            const float g  = 1.0f / (1.0f + expf(-d));
            const float emb_n = fmaxf(sqrtf(nrm[j]), EPS);
            const float cosv  = dot[j] / (emb_n * s_men[c[j]]);
            atomicAdd(&s_intra[c[j]], g * (1.0f - cosv));
        }
    }
    // scalar tail
    for (int n = NV * 4 + gid; n < N; n += gstride) {
        const int c = (int)seg[n];
        const float dx = pts[(size_t)n * 3 + 0] - s_mpts[c * 3 + 0];
        const float dy = pts[(size_t)n * 3 + 1] - s_mpts[c * 3 + 1];
        const float dz = pts[(size_t)n * 3 + 2] - s_mpts[c * 3 + 2];
        const float d  = sqrtf(dx * dx + dy * dy + dz * dz);
        const float g  = 1.0f / (1.0f + expf(-d));
        float dot = 0.0f, nrm = 0.0f;
        for (int k = 0; k < KD; ++k) {
            const float e = emb[(size_t)k * N + n];
            dot += e * s_memb[c * KD + k];
            nrm += e * e;
        }
        const float emb_n = fmaxf(sqrtf(nrm), EPS);
        const float cosv  = dot / (emb_n * s_men[c]);
        atomicAdd(&s_intra[c], g * (1.0f - cosv));
    }
    __syncthreads();
    if (t < 16) atomicAdd(&ws[OFF_INTRA + t], s_intra[t]);
}

// ---------------- kernel 4: WMMA gram (inter) + final combine ----------------
// One full wave32, EXEC all ones at the WMMA. Computes G = mn * mn^T with a
// single v_wmma_f32_16x16x32_f16 (class-0 row of mn is zero, so
// sum(G) - trace(G) over 16x16 equals the reference's 15x15 result).
__global__ __launch_bounds__(32) void k_final(const float* __restrict__ ws,
                                              float* __restrict__ out) {
    const int lane = threadIdx.x;            // 0..31
    const float* mn = ws + OFF_MN;           // [16][32]

    // ---- A fragment: 16-bit A-matrix 16x32 layout (ISA 7.12.2) ----
    const int m  = lane & 15;
    const int kb = (lane < 16) ? 0 : 8;
    v16h a;
#pragma unroll
    for (int i = 0; i < 4; ++i) {
        const int K = kb + 2 * i;
        a[2 * i]     = (_Float16)mn[m * KD + K];
        a[2 * i + 1] = (_Float16)mn[m * KD + K + 1];
    }
#pragma unroll
    for (int i = 0; i < 4; ++i) {
        const int K = kb + 16 + 2 * i;
        a[8 + 2 * i] = (_Float16)mn[m * KD + K];
        a[9 + 2 * i] = (_Float16)mn[m * KD + K + 1];
    }

    // ---- B fragment: 32x16 (K x N), B[k][n] = mn[n][k]; N = lane&15, element == K (same order as A) ----
    v16h b;
#pragma unroll
    for (int e = 0; e < 16; ++e) b[e] = (_Float16)mn[m * KD + kb + (e < 8 ? e : e + 8)];

    v8f c = {};
    c = __builtin_amdgcn_wmma_f32_16x16x32_f16(
            /*neg_a=*/false, a, /*neg_b=*/false, b,
            /*c_mod=*/(short)0, c, /*reuse_a=*/false, /*reuse_b=*/false);

    // C/D layout: lanes 0-15 rows M=r, lanes 16-31 M=r+8; N = lane&15
    const int col = lane & 15;
    const int rb  = (lane < 16) ? 0 : 8;
    float inter_part = 0.0f;
#pragma unroll
    for (int r = 0; r < 8; ++r) {
        const int row = r + rb;
        if (row != col) inter_part += c[r];   // off-diagonal only (sum - trace)
    }

    float intra_part = 0.0f;
    if (lane >= 1 && lane < MC) {
        intra_part = ws[OFF_INTRA + lane] / fmaxf(ws[OFF_CNT + lane], 1.0f);
    }

    __shared__ float red[64];
    red[lane]      = inter_part;
    red[32 + lane] = intra_part;
    __syncthreads();
    if (lane == 0) {
        float inter = 0.0f, intra = 0.0f;
        for (int i = 0; i < 32; ++i) { inter += red[i]; intra += red[32 + i]; }
        out[0] = intra / (float)MC + inter / (float)(MC * (MC - 1));
    }
}

// ---------------- host launcher ----------------
extern "C" void kernel_launch(void* const* d_in, const int* in_sizes, int n_in,
                              void* d_out, int out_size, void* d_ws, size_t ws_size,
                              hipStream_t stream) {
    const float*     pts = (const float*)d_in[0];      // (1, N, 3) f32
    const float*     emb = (const float*)d_in[1];      // (1, K, N) f32
    const long long* seg = (const long long*)d_in[2];  // (1, N) int64
    float* ws  = (float*)d_ws;
    float* out = (float*)d_out;

    const int N  = in_sizes[0] / 3;
    const int NV = N / 4;

    const int threads = 256;                 // 8 wave32s per block
    int blocks = (NV + threads - 1) / threads;
    if (blocks > 2048) blocks = 2048;        // grid-stride; bounds global atomic flushes
    if (blocks < 1) blocks = 1;

    k_zero <<<1, 256, 0, stream>>>(ws);
    k_pass1<<<blocks, threads, 0, stream>>>(pts, emb, seg, ws, N);
    k_means<<<1, 512, 0, stream>>>(ws);
    k_pass2<<<blocks, threads, 0, stream>>>(pts, emb, seg, ws, N);
    k_final<<<1, 32, 0, stream>>>(ws, out);
}
